// Attention_34402688041077
// MI455X (gfx1250) — matrix operand order repaired and checked by reading, not compiled, not run on hardware
//
#include <hip/hip_runtime.h>
#include <hip/hip_bf16.h>

// Problem constants (match reference)
#define B_SZ   2
#define N_SEQ  2048
#define DIM    1024
#define H      16
#define HD     64               // head dim
#define BN     (B_SZ * N_SEQ)   // 4096 rows of x

typedef __bf16 bf16_t;
typedef bf16_t v8bf  __attribute__((ext_vector_type(8)));
typedef bf16_t v16bf __attribute__((ext_vector_type(16)));
typedef float  v8f   __attribute__((ext_vector_type(8)));

// ---------------------------------------------------------------------------
// WMMA fragment loaders (CDNA5 wave32 layouts, CDNA5 ISA §7.12.2)
// ---------------------------------------------------------------------------

// A-matrix 16x32 bf16: lane L -> row L%16; kh = L/16.
// VGPRs 0..3 hold K = kh*8 .. kh*8+7 ; VGPRs 4..7 hold K = 16+kh*8 .. +7.
// p points at (row0, k0) of a row-major [rows, ld] bf16 matrix.
__device__ __forceinline__ v16bf load_a_frag(const bf16_t* p, int ld) {
  const int lane = threadIdx.x & 31;
  const int row  = lane & 15;
  const int kh   = lane >> 4;
  const bf16_t* base = p + (size_t)row * ld;
  v8bf lo = *(const v8bf*)(base + kh * 8);
  v8bf hi = *(const v8bf*)(base + 16 + kh * 8);
  v16bf r;
#pragma unroll
  for (int i = 0; i < 8; ++i) { r[i] = lo[i]; r[i + 8] = hi[i]; }
  return r;
}

// B-matrix 32x16 bf16 (B[k][n]): lane L -> column n = L%16; kh = L/16.
// VGPRs 0..3 hold K = kh*8 .. kh*8+7 ; VGPRs 4..7 hold K = 16+kh*8 .. +7.
// For D = A*B with B[k][n] = Wsrc[n][k], p points at (n0, k0) of row-major
// Wsrc with row stride ld -> lane reads 8 bf16 at k0+kh*8 and 8 at k0+16+kh*8 of row n.
__device__ __forceinline__ v16bf load_b_frag(const bf16_t* p, int ld) {
  const int lane = threadIdx.x & 31;
  const int col  = lane & 15;
  const int kh   = lane >> 4;
  const bf16_t* base = p + (size_t)col * ld;
  v8bf lo = *(const v8bf*)(base + kh * 8);
  v8bf hi = *(const v8bf*)(base + 16 + kh * 8);
  v16bf r;
#pragma unroll
  for (int i = 0; i < 8; ++i) { r[i] = lo[i]; r[i + 8] = hi[i]; }
  return r;
}

__device__ __forceinline__ v8f wmma_bf16(v16bf a, v16bf b, v8f c) {
  return __builtin_amdgcn_wmma_f32_16x16x32_bf16(
      /*neg_a=*/false, a, /*neg_b=*/false, b,
      /*c_mod=*/(short)0, c, /*reuse_a=*/false, /*reuse_b=*/false);
}

// One GEMM pipeline stage: 32(M) x 64(N) x 32(K) tile of fragments.
struct GemmTile {
  v16bf a0, a1;
  v16bf b[4];
};

__device__ __forceinline__ void load_tile(GemmTile& T, const bf16_t* Ap0,
                                          const bf16_t* Ap1, const bf16_t* Wn0,
                                          int k) {
  T.a0 = load_a_frag(Ap0 + k, DIM);
  T.a1 = load_a_frag(Ap1 + k, DIM);
#pragma unroll
  for (int t = 0; t < 4; ++t)
    T.b[t] = load_b_frag(Wn0 + (size_t)t * 16 * DIM + k, DIM);
}

__device__ __forceinline__ void mma_tile(v8f acc[2][4], const GemmTile& T) {
#pragma unroll
  for (int t = 0; t < 4; ++t) {
    acc[0][t] = wmma_bf16(T.a0, T.b[t], acc[0][t]);
    acc[1][t] = wmma_bf16(T.a1, T.b[t], acc[1][t]);
  }
}

// ---------------------------------------------------------------------------
// fp32 -> bf16 conversion
// ---------------------------------------------------------------------------
__global__ void cvt_bf16_kernel(const float* __restrict__ in,
                                bf16_t* __restrict__ out, int n) {
  int i = blockIdx.x * blockDim.x + threadIdx.x;
  if (i < n) out[i] = (bf16_t)in[i];
}

// ---------------------------------------------------------------------------
// Fused QKV GEMM: out[m, nn] = x[m,:] . Wqkv[nn,:] + b[nn],  m in [0,4096),
// nn in [0,3072). Scatters Q,K into [B,H,N,HD] and V transposed [B,H,HD,N].
// Block = 128 threads (4 waves); wave owns a 32(M) x 64(N) tile. k-loop is
// unrolled x2 with ping-pong fragment buffers (no register-rotation moves).
// Grid = (BN/128, 3072/64) = (32, 48).
// ---------------------------------------------------------------------------
__global__ void __launch_bounds__(128, 1)
qkv_gemm_kernel(const bf16_t* __restrict__ xb,
                const bf16_t* __restrict__ wqb,
                const float*  __restrict__ bqkv,
                bf16_t* __restrict__ qbuf,
                bf16_t* __restrict__ kbuf,
                bf16_t* __restrict__ vtbuf) {
  const int wave = threadIdx.x >> 5;
  const int lane = threadIdx.x & 31;
  const int m0 = blockIdx.x * 128 + wave * 32;
  const int n0 = blockIdx.y * 64;

  v8f acc[2][4];
#pragma unroll
  for (int a = 0; a < 2; ++a)
#pragma unroll
    for (int t = 0; t < 4; ++t) acc[a][t] = (v8f){};

  const bf16_t* Ap0 = xb + (size_t)m0 * DIM;
  const bf16_t* Ap1 = xb + (size_t)(m0 + 16) * DIM;
  const bf16_t* Wn0 = wqb + (size_t)n0 * DIM;

  GemmTile TA, TB;
  load_tile(TA, Ap0, Ap1, Wn0, 0);

  int kk = 0;
  for (; kk < DIM - 64; kk += 64) {
    load_tile(TB, Ap0, Ap1, Wn0, kk + 32);
    mma_tile(acc, TA);
    load_tile(TA, Ap0, Ap1, Wn0, kk + 64);
    mma_tile(acc, TB);
  }
  load_tile(TB, Ap0, Ap1, Wn0, kk + 32);  // kk == DIM-64 -> k = DIM-32
  mma_tile(acc, TA);
  mma_tile(acc, TB);

  const int laneLo = lane & 15, laneHi = lane >> 4;
#pragma unroll
  for (int a = 0; a < 2; ++a) {
#pragma unroll
    for (int t = 0; t < 4; ++t) {
#pragma unroll
      for (int r = 0; r < 8; ++r) {
        const int m  = m0 + a * 16 + r + laneHi * 8;
        const int nn = n0 + t * 16 + laneLo;
        const float val = acc[a][t][r] + bqkv[nn];
        const int which = nn >> 10;       // 0=q 1=k 2=v
        const int c = nn & (DIM - 1);
        const int h = c >> 6;             // /HD
        const int d = c & (HD - 1);
        const int bb   = m >> 11;         // /N_SEQ
        const int nseq = m & (N_SEQ - 1);
        const bf16_t v = (bf16_t)val;
        if (which == 0) {
          qbuf[(((size_t)bb * H + h) * N_SEQ + nseq) * HD + d] = v;
        } else if (which == 1) {
          kbuf[(((size_t)bb * H + h) * N_SEQ + nseq) * HD + d] = v;
        } else {
          vtbuf[(((size_t)bb * H + h) * HD + d) * N_SEQ + nseq] = v;  // V^T
        }
      }
    }
  }
}

// ---------------------------------------------------------------------------
// Flash attention: one wave owns a 16-row q tile of one (b,h).
// Block = 128 threads (4 waves). Grid = (B*H, N/(16*4)) = (32, 32).
// j-loop unrolled x2 with ping-pong K-fragment buffers; V frags loaded at the
// top of each block so global latency hides under the softmax VALU work.
// ---------------------------------------------------------------------------
__global__ void __launch_bounds__(128, 1)
attn_kernel(const bf16_t* __restrict__ qbuf,
            const bf16_t* __restrict__ kbuf,
            const bf16_t* __restrict__ vtbuf,
            bf16_t* __restrict__ attnb) {
  __shared__ bf16_t Pbuf[4][16 * 32];   // per-wave 16x32 P tile

  const int wave = threadIdx.x >> 5;
  const int lane = threadIdx.x & 31;
  const int bh = blockIdx.x;                      // b*H + h
  const int q0 = blockIdx.y * 64 + wave * 16;

  const bf16_t* Qp = qbuf  + (size_t)bh * N_SEQ * HD;
  const bf16_t* Kp = kbuf  + (size_t)bh * N_SEQ * HD;
  const bf16_t* Vp = vtbuf + (size_t)bh * HD * N_SEQ;  // [HD, N]

  // Q tile A-fragments (16 x 64 = two 16x32 frags), loaded once.
  const v16bf aq0 = load_a_frag(Qp + (size_t)q0 * HD, HD);
  const v16bf aq1 = load_a_frag(Qp + (size_t)q0 * HD + 32, HD);

  v8f oacc[4];
#pragma unroll
  for (int t = 0; t < 4; ++t) oacc[t] = (v8f){};
  float mstate[8], lstate[8];
#pragma unroll
  for (int r = 0; r < 8; ++r) { mstate[r] = -1e30f; lstate[r] = 0.0f; }

  const float scale = 0.125f;  // 1/sqrt(64)
  bf16_t* Pw = &Pbuf[wave][0];

  // K fragments for a 32-wide key block starting at j.
  auto load_k = [&](v16bf (&bk)[4], int j) {
    bk[0] = load_b_frag(Kp + (size_t)j * HD, HD);
    bk[1] = load_b_frag(Kp + (size_t)j * HD + 32, HD);
    bk[2] = load_b_frag(Kp + (size_t)(j + 16) * HD, HD);
    bk[3] = load_b_frag(Kp + (size_t)(j + 16) * HD + 32, HD);
  };

  // Process one 32-wide key block whose K fragments are already resident.
  auto attn_block = [&](int j0, const v16bf (&bk)[4]) {
    // V fragments for this block, issued early (hide under softmax VALU).
    v16bf bv[4];
#pragma unroll
    for (int t = 0; t < 4; ++t)
      bv[t] = load_b_frag(Vp + (size_t)(t * 16) * N_SEQ + j0, N_SEQ);

    // S = Q * K^T  (two 16x16 tiles: cols j0..j0+15 and j0+16..j0+31)
    v8f s0 = (v8f){}, s1 = (v8f){};
    s0 = wmma_bf16(aq0, bk[0], s0);
    s0 = wmma_bf16(aq1, bk[1], s0);
    s1 = wmma_bf16(aq0, bk[2], s1);
    s1 = wmma_bf16(aq1, bk[3], s1);

    // Online softmax per row (row r+laneHi*8 spans 16 lanes of one half).
#pragma unroll
    for (int r = 0; r < 8; ++r) {
      float x0 = s0[r] * scale;
      float x1 = s1[r] * scale;
      float vmax = fmaxf(x0, x1);
#pragma unroll
      for (int mask = 1; mask < 16; mask <<= 1)
        vmax = fmaxf(vmax, __shfl_xor(vmax, mask, 32));
      const float mnew = fmaxf(mstate[r], vmax);
      const float corr = __expf(mstate[r] - mnew);
      const float p0 = __expf(x0 - mnew);
      const float p1 = __expf(x1 - mnew);
      float rs = p0 + p1;
#pragma unroll
      for (int mask = 1; mask < 16; mask <<= 1)
        rs += __shfl_xor(rs, mask, 32);
      lstate[r] = lstate[r] * corr + rs;
      mstate[r] = mnew;
#pragma unroll
      for (int t = 0; t < 4; ++t) oacc[t][r] *= corr;

      // Stash P (C-layout -> LDS row-major 16x32) for A-fragment reload.
      const int row = r + (lane >> 4) * 8;
      Pw[row * 32 + (lane & 15)]      = (bf16_t)p0;
      Pw[row * 32 + 16 + (lane & 15)] = (bf16_t)p1;
    }

    __builtin_amdgcn_wave_barrier();
    asm volatile("s_wait_dscnt 0" ::: "memory");  // cross-lane LDS dependency

    const v16bf ap = load_a_frag(Pw, 32);  // P as 16x32 A-fragment
    __builtin_amdgcn_wave_barrier();

    // O += P * V   (bv already resident)
#pragma unroll
    for (int t = 0; t < 4; ++t)
      oacc[t] = wmma_bf16(ap, bv[t], oacc[t]);
  };

  v16bf ka[4], kb[4];
  load_k(ka, 0);
  for (int j0 = 0; j0 < N_SEQ; j0 += 64) {
    load_k(kb, j0 + 32);      // issued before block A's softmax
    attn_block(j0, ka);
    // Final iteration reads one tile past kbuf -- lands inside the workspace
    // (vtbuf follows kbuf) and is never consumed.
    load_k(ka, j0 + 64);
    attn_block(j0 + 32, kb);
  }

  // Epilogue: normalize and store to [B, N, DIM] bf16.
  const int h  = bh & (H - 1);
  const int bb = bh >> 4;
#pragma unroll
  for (int t = 0; t < 4; ++t) {
#pragma unroll
    for (int r = 0; r < 8; ++r) {
      const int row = r + (lane >> 4) * 8;
      const float val = oacc[t][r] / lstate[r];
      const int m   = q0 + row;
      const int col = h * HD + t * 16 + (lane & 15);
      attnb[((size_t)bb * N_SEQ + m) * DIM + col] = (bf16_t)val;
    }
  }
}

// ---------------------------------------------------------------------------
// Output projection: out[m, nn] = attn[m,:] . Wproj[nn,:] + bproj[nn] (fp32)
// Grid = (BN/128, DIM/64) = (32, 16). Block = 128 (4 waves, 32x64 per wave).
// Same ping-pong pipeline as the QKV GEMM.
// ---------------------------------------------------------------------------
__global__ void __launch_bounds__(128, 1)
proj_gemm_kernel(const bf16_t* __restrict__ attnb,
                 const bf16_t* __restrict__ wpb,
                 const float*  __restrict__ bproj,
                 float* __restrict__ out) {
  const int wave = threadIdx.x >> 5;
  const int lane = threadIdx.x & 31;
  const int m0 = blockIdx.x * 128 + wave * 32;
  const int n0 = blockIdx.y * 64;

  v8f acc[2][4];
#pragma unroll
  for (int a = 0; a < 2; ++a)
#pragma unroll
    for (int t = 0; t < 4; ++t) acc[a][t] = (v8f){};

  const bf16_t* Ap0 = attnb + (size_t)m0 * DIM;
  const bf16_t* Ap1 = attnb + (size_t)(m0 + 16) * DIM;
  const bf16_t* Wn0 = wpb + (size_t)n0 * DIM;

  GemmTile TA, TB;
  load_tile(TA, Ap0, Ap1, Wn0, 0);

  int kk = 0;
  for (; kk < DIM - 64; kk += 64) {
    load_tile(TB, Ap0, Ap1, Wn0, kk + 32);
    mma_tile(acc, TA);
    load_tile(TA, Ap0, Ap1, Wn0, kk + 64);
    mma_tile(acc, TB);
  }
  load_tile(TB, Ap0, Ap1, Wn0, kk + 32);
  mma_tile(acc, TA);
  mma_tile(acc, TB);

  const int laneLo = lane & 15, laneHi = lane >> 4;
#pragma unroll
  for (int a = 0; a < 2; ++a) {
#pragma unroll
    for (int t = 0; t < 4; ++t) {
#pragma unroll
      for (int r = 0; r < 8; ++r) {
        const int m  = m0 + a * 16 + r + laneHi * 8;
        const int nn = n0 + t * 16 + laneLo;
        out[(size_t)m * DIM + nn] = acc[a][t][r] + bproj[nn];
      }
    }
  }
}

// ---------------------------------------------------------------------------
// Launch
// ---------------------------------------------------------------------------
extern "C" void kernel_launch(void* const* d_in, const int* in_sizes, int n_in,
                              void* d_out, int out_size, void* d_ws, size_t ws_size,
                              hipStream_t stream) {
  const float* x     = (const float*)d_in[0];   // [B,N,DIM]
  const float* Wqkv  = (const float*)d_in[1];   // [3*DIM, DIM]
  const float* bqkv  = (const float*)d_in[2];   // [3*DIM]
  const float* Wproj = (const float*)d_in[3];   // [DIM, DIM]
  const float* bproj = (const float*)d_in[4];   // [DIM]
  float* out = (float*)d_out;

  char* ws = (char*)d_ws;
  size_t off = 0;
  bf16_t* xb    = (bf16_t*)(ws + off); off += (size_t)BN * DIM * sizeof(bf16_t);
  bf16_t* wqb   = (bf16_t*)(ws + off); off += (size_t)3 * DIM * DIM * sizeof(bf16_t);
  bf16_t* wpb   = (bf16_t*)(ws + off); off += (size_t)DIM * DIM * sizeof(bf16_t);
  bf16_t* qbuf  = (bf16_t*)(ws + off); off += (size_t)B_SZ * H * N_SEQ * HD * sizeof(bf16_t);
  bf16_t* kbuf  = (bf16_t*)(ws + off); off += (size_t)B_SZ * H * N_SEQ * HD * sizeof(bf16_t);
  bf16_t* vtbuf = (bf16_t*)(ws + off); off += (size_t)B_SZ * H * HD * N_SEQ * sizeof(bf16_t);
  bf16_t* attnb = (bf16_t*)(ws + off); off += (size_t)BN * DIM * sizeof(bf16_t);
  (void)ws_size; (void)n_in; (void)in_sizes; (void)out_size;

  // 1) fp32 -> bf16 converts
  {
    const int nx = BN * DIM;            // 4,194,304
    const int nq = 3 * DIM * DIM;       // 3,145,728
    const int np = DIM * DIM;           // 1,048,576
    cvt_bf16_kernel<<<(nx + 255) / 256, 256, 0, stream>>>(x, xb, nx);
    cvt_bf16_kernel<<<(nq + 255) / 256, 256, 0, stream>>>(Wqkv, wqb, nq);
    cvt_bf16_kernel<<<(np + 255) / 256, 256, 0, stream>>>(Wproj, wpb, np);
  }

  // 2) fused QKV GEMM + head-split scatter (V transposed)
  qkv_gemm_kernel<<<dim3(BN / 128, (3 * DIM) / 64), 128, 0, stream>>>(
      xb, wqb, bqkv, qbuf, kbuf, vtbuf);

  // 3) flash attention
  attn_kernel<<<dim3(B_SZ * H, N_SEQ / 64), 128, 0, stream>>>(
      qbuf, kbuf, vtbuf, attnb);

  // 4) output projection (fp32 out + bias)
  proj_gemm_kernel<<<dim3(BN / 128, DIM / 64), 128, 0, stream>>>(
      attnb, wpb, bproj, out);
}
